// MetricLearningLoss_39891656245551
// MI455X (gfx1250) — matrix operand order repaired and checked by reading, not compiled, not run on hardware
//
#include <hip/hip_runtime.h>
#include <hip/hip_bf16.h>

// ---------------------------------------------------------------------------
// Metric-learning hard-mining triplet loss, fused WMMA implementation.
//   loss = mean_{valid i} relu( max_{lab[j]!=lab[i]} dot(Ei,Ej) - dot(Ei,Epos(i)) + 1 )
// N = 8192, D = 256 (D hardcoded; K-chunks of 32 for v_wmma_f32_16x16x32_bf16).
// fp32 accuracy recovered via bf16 hi/lo split (3 WMMA products per chunk).
// ---------------------------------------------------------------------------

typedef __bf16 v8bf  __attribute__((ext_vector_type(8)));
typedef __bf16 v16bf __attribute__((ext_vector_type(16)));
typedef float  v8f   __attribute__((ext_vector_type(8)));

constexpr int D  = 256;
constexpr int KC = D / 32;       // 8 K-chunks of 32
constexpr float NEG_INF = -3.0e38f;

__device__ __forceinline__ v16bf cat8(v8bf a, v8bf b) {
    return __builtin_shufflevector(a, b, 0,1,2,3,4,5,6,7,8,9,10,11,12,13,14,15);
}

// ---- kernel 1: per-label first & second occurrence (labels in [0,128)) ----
__global__ void k_label_firsts(const int* __restrict__ labels, int N,
                               int* __restrict__ first, int* __restrict__ second) {
    int t = threadIdx.x;                 // one thread per label value, 128 threads
    int f = -1, s = -1;
    for (int j = 0; j < N; ++j) {
        if (labels[j] == t) {
            if (f < 0)      f = j;
            else            { s = j; break; }
        }
    }
    first[t] = f;
    second[t] = s;
}

// ---- kernel 2: pos_idx + valid per row ------------------------------------
__global__ void k_posidx(const int* __restrict__ labels,
                         const int* __restrict__ first, const int* __restrict__ second,
                         int N, int* __restrict__ pos_idx, int* __restrict__ valid) {
    int i = blockIdx.x * blockDim.x + threadIdx.x;
    if (i >= N) return;
    int L = labels[i];
    int f = first[L];
    int p = (f == i) ? second[L] : f;    // smallest j != i with same label
    int v = (p >= 0) ? 1 : 0;
    if (p < 0) p = i;                    // invalid row: any safe index (masked out)
    pos_idx[i] = p;
    valid[i] = v;
}

// ---- kernel 3: f32 -> bf16 hi/lo split ------------------------------------
__global__ void k_split_bf16(const float* __restrict__ E, long n,
                             __bf16* __restrict__ hi, __bf16* __restrict__ lo) {
    long i = (long)blockIdx.x * blockDim.x + threadIdx.x;
    if (i >= n) return;
    float x = E[i];
    __bf16 h = (__bf16)x;
    hi[i] = h;
    lo[i] = (__bf16)(x - (float)h);
}

// ---- kernel 4: fused Gram + hard mining -----------------------------------
// block = 256 threads = 8 waves; block owns 16 rows, sweeps all N columns.
__global__ void __launch_bounds__(256)
k_mine(const __bf16* __restrict__ hi, const __bf16* __restrict__ lo,
       const int* __restrict__ labels, const int* __restrict__ pos_idx,
       const int* __restrict__ valid, int N, float* __restrict__ margin_out) {
    const int lane = threadIdx.x & 31;
    const int wave = threadIdx.x >> 5;
    const int half = lane >> 4;          // 0: lanes 0-15, 1: lanes 16-31
    const int l15  = lane & 15;
    const int m0   = blockIdx.x * 16;

    // A fragments for rows m0..m0+15, all of K=0..255, hi and lo (kept in VGPRs).
    // A layout: lane half h holds K in [8h, 8h+8) and [8h+16, 8h+24) per 32-chunk.
    v16bf a_hi[KC], a_lo[KC];
    {
        const __bf16* ph = hi + (size_t)(m0 + l15) * D;
        const __bf16* pl = lo + (size_t)(m0 + l15) * D;
#pragma unroll
        for (int kc = 0; kc < KC; ++kc) {
            const int kb = kc * 32 + half * 8;
            a_hi[kc] = cat8(*(const v8bf*)(ph + kb), *(const v8bf*)(ph + kb + 16));
            a_lo[kc] = cat8(*(const v8bf*)(pl + kb), *(const v8bf*)(pl + kb + 16));
        }
    }

    // Per-lane row metadata: C/D VGPR r maps to row m0 + half*8 + r.
    int rowLab[8], rowPos[8];
#pragma unroll
    for (int r = 0; r < 8; ++r) {
        const int row = m0 + half * 8 + r;
        rowLab[r] = labels[row];
        rowPos[r] = pos_idx[row];
    }

    float rowmax[8], posv[8];
#pragma unroll
    for (int r = 0; r < 8; ++r) { rowmax[r] = NEG_INF; posv[r] = NEG_INF; }

    const int tilesPerWave = N >> 7;     // N / (8 waves * 16 cols)
    for (int t = 0; t < tilesPerWave; ++t) {
        const int n0  = (t * 8 + wave) * 16;
        const int col = n0 + l15;
        const __bf16* ch = hi + (size_t)col * D;
        const __bf16* cl = lo + (size_t)col * D;

        v8f acc = {};
#pragma unroll
        for (int kc = 0; kc < KC; ++kc) {
            // B layout: lane half h holds K in [8h, 8h+8) and [8h+16, 8h+24) of the 32-chunk.
            const int kb = kc * 32 + half * 8;
            v16bf b_hi = cat8(*(const v8bf*)(ch + kb), *(const v8bf*)(ch + kb + 16));
            v16bf b_lo = cat8(*(const v8bf*)(cl + kb), *(const v8bf*)(cl + kb + 16));
            // bf16x3 split-product: Ahi*Bhi + Ahi*Blo + Alo*Bhi  (~fp32 accurate)
            acc = __builtin_amdgcn_wmma_f32_16x16x32_bf16(false, a_hi[kc], false, b_hi,
                                                          (short)0, acc, false, false);
            acc = __builtin_amdgcn_wmma_f32_16x16x32_bf16(false, a_hi[kc], false, b_lo,
                                                          (short)0, acc, false, false);
            acc = __builtin_amdgcn_wmma_f32_16x16x32_bf16(false, a_lo[kc], false, b_hi,
                                                          (short)0, acc, false, false);
        }

        // Fused epilogue: hard-negative max over different-label cols + pos gather.
        const int colLab = labels[col];
#pragma unroll
        for (int r = 0; r < 8; ++r) {
            const float v = acc[r];
            if (colLab != rowLab[r]) rowmax[r] = fmaxf(rowmax[r], v);
            if (col == rowPos[r])    posv[r]   = v;
        }
    }

    // Reduce across the 16 lanes of each half (they hold different cols, same row).
#pragma unroll
    for (int off = 1; off < 16; off <<= 1) {
#pragma unroll
        for (int r = 0; r < 8; ++r) {
            rowmax[r] = fmaxf(rowmax[r], __shfl_xor(rowmax[r], off, 32));
            posv[r]   = fmaxf(posv[r],   __shfl_xor(posv[r],   off, 32));
        }
    }

    // Cross-wave combine in LDS (each wave covered a disjoint column subset).
    __shared__ float smax[8][16];
    __shared__ float spos[8][16];
    if (l15 == 0) {
#pragma unroll
        for (int r = 0; r < 8; ++r) {
            smax[wave][half * 8 + r] = rowmax[r];
            spos[wave][half * 8 + r] = posv[r];
        }
    }
    __syncthreads();

    if (threadIdx.x < 16) {
        float m = smax[0][threadIdx.x];
        float p = spos[0][threadIdx.x];
#pragma unroll
        for (int wv = 1; wv < 8; ++wv) {
            m = fmaxf(m, smax[wv][threadIdx.x]);
            p = fmaxf(p, spos[wv][threadIdx.x]);
        }
        const int row = m0 + threadIdx.x;
        // margin = relu(pos_d - neg_d + 1) = relu(negmax_dot - pos_dot + 1)
        const float marg = fmaxf(0.0f, m - p + 1.0f);
        margin_out[row] = valid[row] ? marg : 0.0f;
    }
}

// ---- kernel 5: deterministic final reduction ------------------------------
__global__ void k_finalize(const float* __restrict__ margin,
                           const int* __restrict__ valid, int N,
                           float* __restrict__ out) {
    __shared__ float ssum[256];
    __shared__ float scnt[256];
    const int t = threadIdx.x;
    float s = 0.0f, c = 0.0f;
    for (int i = t; i < N; i += 256) { s += margin[i]; c += (float)valid[i]; }
    ssum[t] = s; scnt[t] = c;
    __syncthreads();
    for (int off = 128; off > 0; off >>= 1) {
        if (t < off) { ssum[t] += ssum[t + off]; scnt[t] += scnt[t + off]; }
        __syncthreads();
    }
    if (t == 0) out[0] = ssum[0] / scnt[0];
}

// ---------------------------------------------------------------------------
extern "C" void kernel_launch(void* const* d_in, const int* in_sizes, int n_in,
                              void* d_out, int out_size, void* d_ws, size_t ws_size,
                              hipStream_t stream) {
    const float* E      = (const float*)d_in[0];
    const int*   labels = (const int*)d_in[1];
    const int N = in_sizes[1];                    // 8192

    // Workspace layout (all 16B-aligned slices).
    char* ws = (char*)d_ws;
    size_t off = 0;
    int* first   = (int*)(ws + off); off += 128 * sizeof(int);
    int* second  = (int*)(ws + off); off += 128 * sizeof(int);
    int* pos_idx = (int*)(ws + off); off += (size_t)N * sizeof(int);
    int* valid   = (int*)(ws + off); off += (size_t)N * sizeof(int);
    float* margin = (float*)(ws + off); off += (size_t)N * sizeof(float);
    off = (off + 127) & ~(size_t)127;
    __bf16* hi = (__bf16*)(ws + off); off += (size_t)N * D * sizeof(__bf16);
    __bf16* lo = (__bf16*)(ws + off); off += (size_t)N * D * sizeof(__bf16);

    k_label_firsts<<<1, 128, 0, stream>>>(labels, N, first, second);
    k_posidx<<<(N + 255) / 256, 256, 0, stream>>>(labels, first, second, N,
                                                  pos_idx, valid);
    const long total = (long)N * D;
    k_split_bf16<<<(int)((total + 255) / 256), 256, 0, stream>>>(E, total, hi, lo);
    k_mine<<<N / 16, 256, 0, stream>>>(hi, lo, labels, pos_idx, valid, N, margin);
    k_finalize<<<1, 256, 0, stream>>>(margin, valid, N, (float*)d_out);
}
